// GatNet_41910290874368
// MI455X (gfx1250) — matrix operand order repaired and checked by reading, not compiled, not run on hardware
//
#include <hip/hip_runtime.h>

// ---------------------------------------------------------------------------
// GAT 2-layer forward for MI455X (gfx1250, wave32, WMMA bf16 16x16x32)
// Block tile 128x128, 8 waves, wave tile 64x32 (4 M x 2 N WMMA subtiles)
// ---------------------------------------------------------------------------
#define N_IN   768
#define N_FEAT 256      // FR == HID == 256, all GEMM outputs are 256 cols
#define N_CLS  4
#define ALPHA  0.2f

#define MT 128
#define NT 128
#define KT 32

typedef __attribute__((ext_vector_type(16))) __bf16 v16bf;
typedef __attribute__((ext_vector_type(8)))  __bf16 v8bf;
typedef __attribute__((ext_vector_type(8)))  float  v8f;

__device__ __forceinline__ unsigned short f32_to_bf16(float f) {
    union { float f; unsigned int u; } v; v.f = f;
    unsigned int u = v.u;
    u += 0x7FFFu + ((u >> 16) & 1u);          // round to nearest even
    return (unsigned short)(u >> 16);
}

// Elementwise convert weight W[Nout, K] (f32 row-major) to bf16, SAME layout.
// W row-major [n][k] is already the column-major-per-output layout the GEMM's
// LDS B tile wants, so no transpose is needed anywhere.
__global__ void convert_w_kernel(const float* __restrict__ W,
                                 unsigned short* __restrict__ Bw, int total) {
    int t = blockIdx.x * blockDim.x + threadIdx.x;
    if (t >= total) return;
    Bw[t] = f32_to_bf16(W[t]);
}

// C[M,256] = A[M,K](f32) x W[256,K](bf16 row-major), f32 accum via WMMA.
__global__ __launch_bounds__(256)
void gemm_bf16_wmma_kernel(const float* __restrict__ A,
                           const unsigned short* __restrict__ Bw,
                           float* __restrict__ C, int M, int K) {
    __shared__ unsigned short lds_a[MT][KT + 8];   // [row][k]  (80B stride)
    __shared__ unsigned short lds_bt[NT][KT + 8];  // [col][k]

    const int tid  = threadIdx.x;
    const int lane = tid & 31;
    const int wave = tid >> 5;
    const int mBase = blockIdx.x * MT;
    const int nBase = blockIdx.y * NT;
    const int wm  = (wave >> 2) * 64;   // 0 / 64
    const int wn  = (wave & 3) * 32;    // 0,32,64,96
    const int hl  = lane >> 4;          // half-wave select
    const int l15 = lane & 15;

    v8f acc[4][2] = {};

    for (int k0 = 0; k0 < K; k0 += KT) {
        // ---- stage A tile 128x32 f32 -> bf16 (float4 per thread x4) ----
        // Row index CLAMPED (not predicated): tail rows re-load row M-1,
        // whose products land only in store-guarded output rows.
        #pragma unroll
        for (int i = 0; i < 4; ++i) {
            int lin = tid + i * 256;            // float4 units, 0..1023
            int r   = lin >> 3;                 // 8 float4 per row
            int c4  = (lin & 7) << 2;
            int gr  = mBase + r; if (gr >= M) gr = M - 1;
            const float* ap = A + (size_t)gr * K + k0 + c4;
            float4 v = *(const float4*)ap;
            __builtin_prefetch(ap + KT, 0, 3);  // speculative global_prefetch_b8
            lds_a[r][c4 + 0] = f32_to_bf16(v.x);
            lds_a[r][c4 + 1] = f32_to_bf16(v.y);
            lds_a[r][c4 + 2] = f32_to_bf16(v.z);
            lds_a[r][c4 + 3] = f32_to_bf16(v.w);
        }
        // ---- stage B tile: 128 cols x 32 k, contiguous b128 copies ----
        #pragma unroll
        for (int i = 0; i < 2; ++i) {
            int lin = tid + i * 256;            // uint4 units, 0..511
            int n   = lin >> 2;                 // col (4 x 16B per col-row)
            int q   = lin & 3;
            uint4 v = *(const uint4*)(Bw + (size_t)(nBase + n) * K + k0 + q * 8);
            *(uint4*)&lds_bt[n][q * 8] = v;
        }
        __syncthreads();

        // ---- B fragments: lane = column, halves hold K 0-15 / 16-31 ----
        v16bf bf[2];
        #pragma unroll
        for (int t = 0; t < 2; ++t) {
            const unsigned short* bp = &lds_bt[wn + t * 16 + l15][0];
            v8bf blo = *(const v8bf*)(bp + (hl ? 16 : 0));
            v8bf bhi = *(const v8bf*)(bp + (hl ? 24 : 8));
            bf[t] = __builtin_shufflevector(blo, bhi,
                        0,1,2,3,4,5,6,7,8,9,10,11,12,13,14,15);
        }

        // ---- 4 M-subtiles x 2 N-subtiles: 8 WMMAs per k-step ----
        #pragma unroll
        for (int s = 0; s < 4; ++s) {
            int row = wm + s * 16 + l15;
            int ka  = hl * 16;
            v8bf alo = *(const v8bf*)&lds_a[row][ka];        // K ka..ka+7
            v8bf ahi = *(const v8bf*)&lds_a[row][ka + 8];    // K ka+8..ka+15
            v16bf af = __builtin_shufflevector(alo, ahi,
                          0,1,2,3,4,5,6,7,8,9,10,11,12,13,14,15);
            acc[s][0] = __builtin_amdgcn_wmma_f32_16x16x32_bf16(
                            false, af, false, bf[0], (short)0, acc[s][0], false, false);
            acc[s][1] = __builtin_amdgcn_wmma_f32_16x16x32_bf16(
                            false, af, false, bf[1], (short)0, acc[s][1], false, false);
        }
        __syncthreads();
    }

    // ---- store C: VGPR v -> M = v + 8*hl, N = l15 ----
    if (mBase + MT <= M) {
        // full tile (common case): straight-line stores, no guards
        #pragma unroll
        for (int s = 0; s < 4; ++s)
            #pragma unroll
            for (int t = 0; t < 2; ++t)
                #pragma unroll
                for (int v = 0; v < 8; ++v) {
                    int gr = mBase + wm + s * 16 + v + hl * 8;
                    int gc = nBase + wn + t * 16 + l15;
                    C[(size_t)gr * N_FEAT + gc] = acc[s][t][v];
                }
    } else {
        #pragma unroll
        for (int s = 0; s < 4; ++s)
            #pragma unroll
            for (int t = 0; t < 2; ++t)
                #pragma unroll
                for (int v = 0; v < 8; ++v) {
                    int gr = mBase + wm + s * 16 + v + hl * 8;
                    int gc = nBase + wn + t * 16 + l15;
                    if (gr < M) C[(size_t)gr * N_FEAT + gc] = acc[s][t][v];
                }
    }
}

// f1[n] = h[n,:] . a1 + b1 ; f2[n] = h[n,:] . a2 + b2    (one wave per node)
__global__ void f1f2_kernel(const float* __restrict__ h,
                            const float* __restrict__ a1, const float* __restrict__ b1,
                            const float* __restrict__ a2, const float* __restrict__ b2,
                            float* __restrict__ f1, float* __restrict__ f2, int N) {
    int w    = (blockIdx.x * blockDim.x + threadIdx.x) >> 5;
    int lane = threadIdx.x & 31;
    if (w >= N) return;
    const float4* hv  = (const float4*)(h + (size_t)w * N_FEAT);
    const float4* w1v = (const float4*)a1;
    const float4* w2v = (const float4*)a2;
    float s1 = 0.f, s2 = 0.f;
    #pragma unroll
    for (int j = 0; j < 2; ++j) {
        float4 v  = hv[lane * 2 + j];
        float4 u1 = w1v[lane * 2 + j];
        float4 u2 = w2v[lane * 2 + j];
        s1 += v.x*u1.x + v.y*u1.y + v.z*u1.z + v.w*u1.w;
        s2 += v.x*u2.x + v.y*u2.y + v.z*u2.z + v.w*u2.w;
    }
    #pragma unroll
    for (int off = 16; off > 0; off >>= 1) {
        s1 += __shfl_down(s1, off, 32);
        s2 += __shfl_down(s2, off, 32);
    }
    if (lane == 0) { f1[w] = s1 + b1[0]; f2[w] = s2 + b2[0]; }
}

__global__ void zero_kernel(float* __restrict__ p, long long n) {
    long long i = (long long)blockIdx.x * blockDim.x + threadIdx.x;
    long long st = (long long)gridDim.x * blockDim.x;
    for (; i < n; i += st) p[i] = 0.f;
}

// coefs[e] = exp(leaky_relu(f1[src] + f2[dst])); csum[src] += coefs[e]
__global__ void edge_coef_kernel(const int* __restrict__ src, const int* __restrict__ dst,
                                 const float* __restrict__ f1, const float* __restrict__ f2,
                                 float* __restrict__ coefs, float* __restrict__ csum, int E) {
    int e = blockIdx.x * blockDim.x + threadIdx.x;
    if (e >= E) return;
    int s = src[e], d = dst[e];
    float l = f1[s] + f2[d];
    l = (l > 0.f) ? l : ALPHA * l;
    float c = __expf(l);
    coefs[e] = c;
    atomicAdd(&csum[s], c);
}

// agg[src,:] += coefs[e] * h[dst,:]   (one wave per edge; h is L2-resident)
__global__ void edge_agg_kernel(const int* __restrict__ src, const int* __restrict__ dst,
                                const float* __restrict__ coefs,
                                const float* __restrict__ h,
                                float* __restrict__ agg, int E) {
    int w    = (blockIdx.x * blockDim.x + threadIdx.x) >> 5;
    int lane = threadIdx.x & 31;
    if (w >= E) return;
    int s = src[w], d = dst[w];
    float c = coefs[w];
    const float4* hv = (const float4*)(h + (size_t)d * N_FEAT);
    float* ag = agg + (size_t)s * N_FEAT;
    #pragma unroll
    for (int j = 0; j < 2; ++j) {
        float4 v = hv[lane * 2 + j];
        int base = (lane * 2 + j) * 4;
        atomicAdd(ag + base + 0, c * v.x);
        atomicAdd(ag + base + 1, c * v.y);
        atomicAdd(ag + base + 2, c * v.z);
        atomicAdd(ag + base + 3, c * v.w);
    }
}

// out[n,c] = agg[n,c] / csum[n] + bias[c]
__global__ void finalize_kernel(const float* __restrict__ agg, const float* __restrict__ csum,
                                const float* __restrict__ bias, float* __restrict__ out,
                                long long total) {
    long long i = (long long)blockIdx.x * blockDim.x + threadIdx.x;
    if (i >= total) return;
    int n = (int)(i >> 8), c = (int)(i & 255);
    out[i] = agg[i] / csum[n] + bias[c];
}

// out[n,c] = x2[n,:] . W_cls[c,:] + b_cls[c]   (one wave per node, 4 classes)
__global__ void classifier_kernel(const float* __restrict__ x2,
                                  const float* __restrict__ Wc, const float* __restrict__ bc,
                                  float* __restrict__ out, int N) {
    int w    = (blockIdx.x * blockDim.x + threadIdx.x) >> 5;
    int lane = threadIdx.x & 31;
    if (w >= N) return;
    const float4* xv = (const float4*)(x2 + (size_t)w * N_FEAT);
    float acc[N_CLS] = {0.f, 0.f, 0.f, 0.f};
    #pragma unroll
    for (int j = 0; j < 2; ++j) {
        float4 v = xv[lane * 2 + j];
        #pragma unroll
        for (int c = 0; c < N_CLS; ++c) {
            float4 u = ((const float4*)(Wc + c * N_FEAT))[lane * 2 + j];
            acc[c] += v.x*u.x + v.y*u.y + v.z*u.z + v.w*u.w;
        }
    }
    #pragma unroll
    for (int c = 0; c < N_CLS; ++c)
        #pragma unroll
        for (int off = 16; off > 0; off >>= 1)
            acc[c] += __shfl_down(acc[c], off, 32);
    if (lane == 0)
        #pragma unroll
        for (int c = 0; c < N_CLS; ++c)
            out[(size_t)w * N_CLS + c] = acc[c] + bc[c];
}

extern "C" void kernel_launch(void* const* d_in, const int* in_sizes, int n_in,
                              void* d_out, int out_size, void* d_ws, size_t ws_size,
                              hipStream_t stream) {
    (void)n_in; (void)out_size; (void)ws_size;
    const float* x      = (const float*)d_in[0];
    const int*   edges  = (const int*)d_in[1];
    const float* W_lin1 = (const float*)d_in[2];
    const float* W_seq1 = (const float*)d_in[3];
    const float* a1_1   = (const float*)d_in[4];
    const float* b1_1   = (const float*)d_in[5];
    const float* a2_1   = (const float*)d_in[6];
    const float* b2_1   = (const float*)d_in[7];
    const float* bias1  = (const float*)d_in[8];
    const float* W_lin2 = (const float*)d_in[9];
    const float* W_seq2 = (const float*)d_in[10];
    const float* a1_2   = (const float*)d_in[11];
    const float* b1_2   = (const float*)d_in[12];
    const float* a2_2   = (const float*)d_in[13];
    const float* b2_2   = (const float*)d_in[14];
    const float* bias2  = (const float*)d_in[15];
    const float* W_cls  = (const float*)d_in[16];
    const float* b_cls  = (const float*)d_in[17];

    const int N = in_sizes[0] / N_IN;     // 50000
    const int E = in_sizes[1] / 2;        // 1600000
    const int* src = edges;
    const int* dst = edges + E;

    // ---------------- workspace layout (f32 unless noted) ----------------
    const size_t nh = (size_t)N * N_FEAT;
    float* fA    = (float*)d_ws;          // GEMM1 output (x @ Wlin.T)
    float* fH    = fA + nh;               // h
    float* fAgg  = fH + nh;               // agg / layer output
    float* f1    = fAgg + nh;
    float* f2    = f1 + N;
    float* csum  = f2 + N;
    float* coefs = csum + N;
    unsigned short* wlin1b = (unsigned short*)(coefs + E);
    unsigned short* wseq1b = wlin1b + (size_t)N_FEAT * N_IN;
    unsigned short* wlin2b = wseq1b + (size_t)N_FEAT * N_FEAT;
    unsigned short* wseq2b = wlin2b + (size_t)N_FEAT * N_FEAT;

    float* out_logits = (float*)d_out;                  // [N, 4]
    float* out_x2     = out_logits + (size_t)N * N_CLS; // [N, 256]

    const dim3 blk(256);
    const dim3 gemm_grid((N + MT - 1) / MT, N_FEAT / NT);   // (391, 2)
    const int  waveBlocks = (N * 32 + 255) / 256;
    const int  aggBlocks  = (E * 32 + 255) / 256;

    // ---- weight bf16 conversion (layout-preserving, cheap) ----
    convert_w_kernel<<<(N_FEAT * N_IN + 255) / 256, blk, 0, stream>>>(W_lin1, wlin1b, N_FEAT * N_IN);
    convert_w_kernel<<<(N_FEAT * N_FEAT + 255) / 256, blk, 0, stream>>>(W_seq1, wseq1b, N_FEAT * N_FEAT);
    convert_w_kernel<<<(N_FEAT * N_FEAT + 255) / 256, blk, 0, stream>>>(W_lin2, wlin2b, N_FEAT * N_FEAT);
    convert_w_kernel<<<(N_FEAT * N_FEAT + 255) / 256, blk, 0, stream>>>(W_seq2, wseq2b, N_FEAT * N_FEAT);

    // ============================ layer 1 ============================
    gemm_bf16_wmma_kernel<<<gemm_grid, blk, 0, stream>>>(x,  wlin1b, fA, N, N_IN);
    gemm_bf16_wmma_kernel<<<gemm_grid, blk, 0, stream>>>(fA, wseq1b, fH, N, N_FEAT);
    f1f2_kernel<<<waveBlocks, blk, 0, stream>>>(fH, a1_1, b1_1, a2_1, b2_1, f1, f2, N);
    zero_kernel<<<2048, blk, 0, stream>>>(csum, N);
    zero_kernel<<<2048, blk, 0, stream>>>(fAgg, (long long)nh);
    edge_coef_kernel<<<(E + 255) / 256, blk, 0, stream>>>(src, dst, f1, f2, coefs, csum, E);
    edge_agg_kernel<<<aggBlocks, blk, 0, stream>>>(src, dst, coefs, fH, fAgg, E);
    finalize_kernel<<<(int)((nh + 255) / 256), blk, 0, stream>>>(fAgg, csum, bias1, fAgg, (long long)nh);

    // ============================ layer 2 ============================
    gemm_bf16_wmma_kernel<<<gemm_grid, blk, 0, stream>>>(fAgg, wlin2b, fA, N, N_FEAT);
    gemm_bf16_wmma_kernel<<<gemm_grid, blk, 0, stream>>>(fA,   wseq2b, fH, N, N_FEAT);
    f1f2_kernel<<<waveBlocks, blk, 0, stream>>>(fH, a1_2, b1_2, a2_2, b2_2, f1, f2, N);
    zero_kernel<<<2048, blk, 0, stream>>>(csum, N);
    zero_kernel<<<2048, blk, 0, stream>>>(fAgg, (long long)nh);   // x1 consumed above
    edge_coef_kernel<<<(E + 255) / 256, blk, 0, stream>>>(src, dst, f1, f2, coefs, csum, E);
    edge_agg_kernel<<<aggBlocks, blk, 0, stream>>>(src, dst, coefs, fH, fAgg, E);
    finalize_kernel<<<(int)((nh + 255) / 256), blk, 0, stream>>>(fAgg, csum, bias2, out_x2, (long long)nh);

    // =========================== classifier ==========================
    classifier_kernel<<<waveBlocks, blk, 0, stream>>>(out_x2, W_cls, b_cls, out_logits, N);
}
